// LaneNodeEmbedding_77489799954822
// MI455X (gfx1250) — matrix operand order repaired and checked by reading, not compiled, not run on hardware
//
#include <hip/hip_runtime.h>
#include <hip/hip_bf16.h>

typedef __attribute__((ext_vector_type(16))) _Float16 v16h;
typedef __attribute__((ext_vector_type(8)))  float    v8f;

#define LN_EPS   1e-5f
#define NORM_EPS 1e-6f

#define NODES_PER_WAVE 4
#define WAVES_PER_BLOCK 8

// ---- workspace layout (bytes) ----
// [0, 16384)        : W2 packed to WMMA B-operand layout, f16
//                     slot t=[nt(8)][ktile(2)][lane(32)] -> 16 contiguous f16
// [16384, 17408)    : stage-1 table float4(w0, w1, b1, gamma) [half(2)][i(32)]
// [17408, 17664)    : stage-1 table beta                      [half(2)][i(32)]
// [17664, 17700)    : 9 LN-statistic scalars (closed-form mean/var of h)
#define WS_OFF_B   0
#define WS_OFF_T4  16384
#define WS_OFF_LB  17408
#define WS_OFF_S   17664

__device__ __forceinline__ int jmap(int i, int half) {
    // lane's hidden-channel set matching the 16x32 f16 A-operand VGPR layout
    return ((i >> 3) << 4) + (i & 7) + half * 8;
}

// ---------------------------------------------------------------------------
// One-off pack kernel (idempotent, graph-capture safe).
// ---------------------------------------------------------------------------
__global__ __launch_bounds__(512)
void pack_kernel(const float* __restrict__ W1,   // [2,64]
                 const float* __restrict__ b1,   // [64]
                 const float* __restrict__ lng,  // [64]
                 const float* __restrict__ lnb,  // [64]
                 const float* __restrict__ W2,   // [64,128]
                 char* __restrict__ ws)
{
    const int t = threadIdx.x;                  // 0..511
    // --- W2 -> B-operand f16 layout ---
    {
        const int lane  = t & 31;
        const int ktile = (t >> 5) & 1;
        const int nt    = t >> 6;
        const int c     = nt * 16 + (lane & 15);
        const int koff  = ktile * 32 + (lane >> 4) * 8;
        _Float16* dst = (_Float16*)(ws + WS_OFF_B) + (size_t)t * 16;
        #pragma unroll
        for (int e = 0; e < 16; ++e)
            dst[e] = (_Float16)W2[(koff + ((e >> 3) << 4) + (e & 7)) * 128 + c];
    }
    // --- stage-1 per-(half,i) tables in A-layout channel order ---
    if (t < 64) {
        const int half = t >> 5, i = t & 31;
        const int j = jmap(i, half);
        float4 w; w.x = W1[j]; w.y = W1[64 + j]; w.z = b1[j]; w.w = lng[j];
        ((float4*)(ws + WS_OFF_T4))[t] = w;
        ((float*) (ws + WS_OFF_LB))[t] = lnb[j];
    }
    // --- closed-form LN statistics: h = dx*w0 + dy*w1 + b is affine in
    //     (dx,dy), so mean(h) and mean(h^2) over 64 channels reduce to 9
    //     precomputed sums. ---
    if (t == 0) {
        float S0=0,S1=0,Sb=0,Q00=0,Q11=0,Qbb=0,Q01=0,Q0b=0,Q1b=0;
        for (int j = 0; j < 64; ++j) {
            float a = W1[j], b = W1[64 + j], c = b1[j];
            S0 += a;   S1 += b;   Sb  += c;
            Q00 += a*a; Q11 += b*b; Qbb += c*c;
            Q01 += a*b; Q0b += a*c; Q1b += b*c;
        }
        float* S = (float*)(ws + WS_OFF_S);
        const float inv = 1.0f / 64.0f;
        S[0] = S0*inv;  S[1] = S1*inv;  S[2] = Sb*inv;
        S[3] = Q00*inv; S[4] = Q11*inv; S[5] = Qbb*inv;
        S[6] = 2.f*Q01*inv; S[7] = 2.f*Q0b*inv; S[8] = 2.f*Q1b*inv;
    }
}

// ---------------------------------------------------------------------------
// Fused main kernel: one wave = 4 lane-nodes, each padded to an M=16 WMMA
// tile (rows 9..15 duplicate point 8 -> harmless under max).
// ---------------------------------------------------------------------------
__global__ __launch_bounds__(256)
void lane_node_embed_kernel(const float* __restrict__ pos,   // [NL,10,2]
                            const int*   __restrict__ idxI,  // [NL]
                            const int*   __restrict__ idxT,  // [NL]
                            const int*   __restrict__ idxC,  // [NL]
                            const char*  __restrict__ ws,
                            const float* __restrict__ b2,    // [128]
                            const float* __restrict__ Ei,    // [3,128]
                            const float* __restrict__ Et,    // [3,128]
                            const float* __restrict__ Ec,    // [2,128]
                            float* __restrict__ out,         // [NL,128]
                            int NL)
{
    const int lane  = threadIdx.x & 31;
    const int wave  = blockIdx.x * WAVES_PER_BLOCK + (threadIdx.x >> 5);
    const int row   = lane & 15;
    const int half  = lane >> 4;
    const int p     = row < 8 ? row : 8;     // pad rows 9..15 with point 8
    const int n0    = wave * NODES_PER_WAVE;

    const v16h*   Bws = (const v16h*)  (ws + WS_OFF_B);
    const float4* T4  = (const float4*)(ws + WS_OFF_T4) + half * 32;
    const float*  LB  = (const float*) (ws + WS_OFF_LB) + half * 32;
    const float*  S   = (const float*) (ws + WS_OFF_S);   // uniform -> s_load
    const float s0 = S[0], s1 = S[1], sb  = S[2];
    const float q00 = S[3], q11 = S[4], qbb = S[5];
    const float q01 = S[6], q0b = S[7], q1b = S[8];

    // ---- Phase 1: dv -> Linear1 + LN(closed-form stats) + ReLU, packed
    //      straight into WMMA A-operand layout. ----
    v16h A0[NODES_PER_WAVE];   // K =  0..31
    v16h A1[NODES_PER_WAVE];   // K = 32..63
    #pragma unroll
    for (int nn = 0; nn < NODES_PER_WAVE; ++nn) {
        int n  = n0 + nn;
        int nc = (n < NL) ? n : 0;
        const float2* pp = (const float2*)(pos + (size_t)nc * 20 + p * 2);
        float2 pa = pp[0], pb = pp[1];
        float dx = pb.x - pa.x, dy = pb.y - pa.y;
        float inv = __frsqrt_rn(dx * dx + dy * dy + NORM_EPS);
        dx *= inv; dy *= inv;

        // closed-form LayerNorm statistics over the 64 hidden channels
        float mu  = fmaf(dx, s0, fmaf(dy, s1, sb));
        float eh2 = fmaf(dx * dx, q00, fmaf(dy * dy, q11, qbb))
                  + fmaf(dx * dy, q01, fmaf(dx, q0b, dy * q1b));
        float var = eh2 - mu * mu;
        float rs  = __frsqrt_rn(var + LN_EPS);
        float nmr = -mu * rs;

        #pragma unroll
        for (int i = 0; i < 32; ++i) {
            float4 w = T4[i];                       // (w0, w1, b1, gamma)
            float h  = fmaf(dx, w.x, fmaf(dy, w.y, w.z));
            float tv = fmaf(h, rs, nmr);            // (h - mu) * rs
            float v  = fmaxf(fmaf(tv, w.w, LB[i]), 0.f);   // *g + beta, ReLU
            _Float16 hv = (_Float16)v;
            if (i < 16) A0[nn][i] = hv; else A1[nn][i - 16] = hv;
        }
    }

    // ---- Phase 2: Linear2 via v_wmma_f32_16x16x32_f16, b2 folded into the
    //      C operand, max over the 9 points inside the C layout. ----
    v8f res[NODES_PER_WAVE];
    const int col16 = lane & 15;
    #pragma unroll
    for (int nt = 0; nt < 8; ++nt) {
        v16h B0 = Bws[(nt * 2 + 0) * 32 + lane];
        v16h B1 = Bws[(nt * 2 + 1) * 32 + lane];
        float b2v = b2[nt * 16 + col16];            // constant per column
        #pragma unroll
        for (int nn = 0; nn < NODES_PER_WAVE; ++nn) {
            v8f acc = {b2v, b2v, b2v, b2v, b2v, b2v, b2v, b2v};
            acc = __builtin_amdgcn_wmma_f32_16x16x32_f16(
                false, A0[nn], false, B0, (short)0, acc, false, false);
            acc = __builtin_amdgcn_wmma_f32_16x16x32_f16(
                false, A1[nn], false, B1, (short)0, acc, false, false);
            // lanes 0-15: rows 0-7; lanes 16-31: rows 8-15 (9..15 dup row 8)
            float m = acc[0];
            #pragma unroll
            for (int gi = 1; gi < 8; ++gi) m = fmaxf(m, acc[gi]);
            m = fmaxf(m, __shfl_xor(m, 16, 32));
            res[nn][nt] = m;
        }
    }

    // ---- Phase 3: + gathered discrete embeddings, store [NL,128]. ----
    if (lane < 16) {
        #pragma unroll
        for (int nn = 0; nn < NODES_PER_WAVE; ++nn) {
            int n = n0 + nn;
            if (n < NL) {
                int ii = idxI[n], it = idxT[n], ic = idxC[n];
                #pragma unroll
                for (int nt = 0; nt < 8; ++nt) {
                    int c = nt * 16 + lane;
                    float d = Ei[ii * 128 + c] + Et[it * 128 + c] + Ec[ic * 128 + c];
                    out[(size_t)n * 128 + c] = res[nn][nt] + d;
                }
            }
        }
    }
}

extern "C" void kernel_launch(void* const* d_in, const int* in_sizes, int n_in,
                              void* d_out, int out_size, void* d_ws, size_t ws_size,
                              hipStream_t stream) {
    const float* pos  = (const float*)d_in[0];
    const int*   idxI = (const int*)  d_in[1];
    const int*   idxT = (const int*)  d_in[2];
    const int*   idxC = (const int*)  d_in[3];
    const float* W1   = (const float*)d_in[4];
    const float* b1   = (const float*)d_in[5];
    const float* lng  = (const float*)d_in[6];
    const float* lnb  = (const float*)d_in[7];
    const float* W2   = (const float*)d_in[8];
    const float* b2   = (const float*)d_in[9];
    const float* Ei   = (const float*)d_in[10];
    const float* Et   = (const float*)d_in[11];
    const float* Ec   = (const float*)d_in[12];
    float* out = (float*)d_out;
    char* ws = (char*)d_ws;

    pack_kernel<<<1, 512, 0, stream>>>(W1, b1, lng, lnb, W2, ws);

    const int NL = in_sizes[1];                                   // B * L
    const int nodesPerBlock = NODES_PER_WAVE * WAVES_PER_BLOCK;   // 32
    const int blocks = (NL + nodesPerBlock - 1) / nodesPerBlock;

    lane_node_embed_kernel<<<blocks, 32 * WAVES_PER_BLOCK, 0, stream>>>(
        pos, idxI, idxT, idxC, ws, b2, Ei, Et, Ec, out, NL);
}
